// FBSNN_7782480740720
// MI455X (gfx1250) — matrix operand order repaired and checked by reading, not compiled, not run on hardware
//
#include <hip/hip_runtime.h>
#include <hip/hip_bf16.h>

typedef __attribute__((ext_vector_type(16))) __bf16 v16bf;
typedef __attribute__((ext_vector_type(8)))  float  v8f;

#define M_TOTAL 4096
#define D_IN    100
#define HIDN    256
#define NCOLS   10000   // D_IN*D_IN flattened Hessian columns
#define NPAD    10112   // 79 * 128
#define NBLK    128
#define MBLK    64
#define LDSK    136     // 128 + 8 pad (bf16) -> 272B row stride, 16B aligned

__device__ __forceinline__ unsigned short f2bf(float f) {
    unsigned int x = __float_as_uint(f);
    x += 0x7FFFu + ((x >> 16) & 1u);          // round-to-nearest-even
    return (unsigned short)(x >> 16);
}

union Frag {
    uint4 q[2];
    unsigned short u[16];
    v16bf v;
};

// ---- Kernel 1: hidden layer: z = W1 s + b1; c = W2*sin(z); gc = W2*cos(z);
//      u = sum(c) + b2. Writes gc (f32), c (bf16 for WMMA A), u.
__global__ void __launch_bounds__(256) fbsnn_hidden(
    const float* __restrict__ t, const float* __restrict__ X,
    const float* __restrict__ W1, const float* __restrict__ b1,
    const float* __restrict__ W2, const float* __restrict__ b2,
    float* __restrict__ gc, unsigned short* __restrict__ Cb,
    float* __restrict__ u_out)
{
    const int s = blockIdx.x;
    const int h = threadIdx.x;           // 256 threads = HIDN
    __shared__ float inb[101];
    __shared__ float red[256];
    for (int d = h; d < 101; d += 256)
        inb[d] = (d == 0) ? t[s] : X[(size_t)s * D_IN + (d - 1)];
    __syncthreads();

    const float* w = W1 + h * 101;
    float z = b1[h];
    #pragma unroll 4
    for (int d = 0; d < 101; ++d) z += inb[d] * w[d];

    const float w2 = W2[h];
    const float c  = w2 * sinf(z);
    const float g  = w2 * cosf(z);
    gc[(size_t)s * HIDN + h] = g;
    Cb[(size_t)s * HIDN + h] = f2bf(c);

    red[h] = c;
    __syncthreads();
    for (int off = 128; off > 0; off >>= 1) {
        if (h < off) red[h] += red[h + off];
        __syncthreads();
    }
    if (h == 0) u_out[s] = red[0] + b2[0];
}

// ---- Kernel 2: gradient g = gc @ W1  (per sample, 101 outputs)
__global__ void __launch_bounds__(128) fbsnn_grad(
    const float* __restrict__ gc, const float* __restrict__ W1,
    float* __restrict__ DuDt, float* __restrict__ DuDx)
{
    const int s = blockIdx.x;
    const int j = threadIdx.x;
    __shared__ float g[HIDN];
    for (int h = j; h < HIDN; h += 128) g[h] = gc[(size_t)s * HIDN + h];
    __syncthreads();
    if (j < 101) {
        float acc = 0.f;
        #pragma unroll 8
        for (int h = 0; h < HIDN; ++h) acc += g[h] * W1[h * 101 + j];
        if (j == 0) DuDt[s] = acc;
        else        DuDx[(size_t)s * D_IN + (j - 1)] = acc;
    }
}

// ---- Kernel 3: PT[n][h] = W1[h,1+j]*W1[h,1+k] (n = j*100+k), bf16, zero-padded
__global__ void __launch_bounds__(256) fbsnn_build_pt(
    const float* __restrict__ W1, unsigned short* __restrict__ PT)
{
    const int idx = blockIdx.x * 256 + threadIdx.x;   // == n*256 + h (coalesced)
    const int h = idx & 255;
    const int n = idx >> 8;
    float v = 0.f;
    if (n < NCOLS) {
        const int j = n / D_IN, k = n % D_IN;
        v = W1[h * 101 + 1 + j] * W1[h * 101 + 1 + k];
    }
    PT[idx] = f2bf(v);
}

// ---- Kernel 4: H[s][n] = -sum_h Cb[s][h] * PT[n][h], bf16 WMMA, f32 accum.
//      Block: 256 threads (8 waves), tile 64(M) x 128(N), K=256 in two LDS stages.
//      Each wave computes a 32x32 sub-tile (4 accumulator fragments).
__global__ void __launch_bounds__(256) fbsnn_hessian_wmma(
    const unsigned short* __restrict__ Cb,    // [4096][256] bf16 row-major
    const unsigned short* __restrict__ PTb,   // [NPAD][256] bf16 row-major
    float* __restrict__ H)                    // [4096][10000] f32
{
    __shared__ unsigned short As[MBLK][LDSK];   // 17.4 KB
    __shared__ unsigned short Bs[NBLK][LDSK];   // 34.8 KB

    const int n0   = blockIdx.x * NBLK;
    const int s0   = blockIdx.y * MBLK;
    const int tid  = threadIdx.x;
    const int lane = tid & 31;
    const int wid  = tid >> 5;
    const int wm   = wid >> 2;        // 0..1  (M direction)
    const int wn   = wid & 3;         // 0..3  (N direction)
    const int l15  = lane & 15;
    const int hi   = lane >> 4;       // half-wave select

    const v8f zero = {0.f,0.f,0.f,0.f,0.f,0.f,0.f,0.f};
    v8f acc[2][2];
    acc[0][0] = zero; acc[0][1] = zero; acc[1][0] = zero; acc[1][1] = zero;

    for (int kk = 0; kk < HIDN; kk += 128) {
        // Stage A tile: 64 rows x 128 K  (16B vector loads, coalesced)
        #pragma unroll
        for (int it = 0; it < 4; ++it) {
            const int li = tid + it * 256;         // 0..1023
            const int r  = li >> 4;                // 16 uint4 per row
            const int c  = (li & 15) << 3;
            *(uint4*)&As[r][c] =
                *(const uint4*)&Cb[((size_t)(s0 + r)) * HIDN + kk + c];
        }
        // Stage B tile: 128 rows x 128 K
        #pragma unroll
        for (int it = 0; it < 8; ++it) {
            const int li = tid + it * 256;         // 0..2047
            const int r  = li >> 4;
            const int c  = (li & 15) << 3;
            *(uint4*)&Bs[r][c] =
                *(const uint4*)&PTb[((size_t)(n0 + r)) * HIDN + kk + c];
        }
        __syncthreads();

        #pragma unroll
        for (int k0 = 0; k0 < 128; k0 += 32) {
            Frag a[2], b[2];
            const int ka   = k0 + hi * 8;    // A: half-waves split K by 0/+8, +16 block
            const int kb16 = k0 + hi * 8;    // B: half-waves split K by 0/+8, +16 block
            #pragma unroll
            for (int i = 0; i < 2; ++i) {
                const int m = wm * 32 + i * 16 + l15;   // row per lane (ISA 7.12.2)
                a[i].q[0] = *(const uint4*)&As[m][ka];       // K = ka .. ka+7
                a[i].q[1] = *(const uint4*)&As[m][ka + 16];  // K = ka+16 .. ka+23
            }
            #pragma unroll
            for (int j = 0; j < 2; ++j) {
                const int n = wn * 32 + j * 16 + l15;   // col per lane
                b[j].q[0] = *(const uint4*)&Bs[n][kb16];
                b[j].q[1] = *(const uint4*)&Bs[n][kb16 + 16];
            }
            #pragma unroll
            for (int i = 0; i < 2; ++i)
                #pragma unroll
                for (int j = 0; j < 2; ++j)
                    acc[i][j] = __builtin_amdgcn_wmma_f32_16x16x32_bf16(
                        false, a[i].v, false, b[j].v,
                        (short)0, acc[i][j], false, false);
        }
        __syncthreads();
    }

    // D layout: col = lane&15; row = v + 8*(lane>=16). Negate on store.
    const int rhalf = hi << 3;
    #pragma unroll
    for (int i = 0; i < 2; ++i) {
        #pragma unroll
        for (int j = 0; j < 2; ++j) {
            const int ncol = n0 + wn * 32 + j * 16 + l15;
            if (ncol < NCOLS) {
                #pragma unroll
                for (int v = 0; v < 8; ++v) {
                    const int srow = s0 + wm * 32 + i * 16 + v + rhalf;
                    H[(size_t)srow * NCOLS + ncol] = -acc[i][j][v];
                }
            }
        }
    }
}

extern "C" void kernel_launch(void* const* d_in, const int* in_sizes, int n_in,
                              void* d_out, int out_size, void* d_ws, size_t ws_size,
                              hipStream_t stream) {
    const float* t  = (const float*)d_in[0];
    const float* X  = (const float*)d_in[1];
    const float* W1 = (const float*)d_in[2];
    const float* b1 = (const float*)d_in[3];
    const float* W2 = (const float*)d_in[4];
    const float* b2 = (const float*)d_in[5];

    float* out  = (float*)d_out;                       // tuple order: u, DuDx, DuDt, D2uDx2
    float* u    = out;                                 // 4096
    float* DuDx = out + M_TOTAL;                       // 4096*100
    float* DuDt = DuDx + (size_t)M_TOTAL * D_IN;       // 4096
    float* H    = DuDt + M_TOTAL;                      // 4096*100*100

    char* ws = (char*)d_ws;
    float*          gc  = (float*)ws;                          // 4 MB
    unsigned short* Cb  = (unsigned short*)(ws + (4u << 20));  // 2 MB (bf16 A)
    unsigned short* PTb = (unsigned short*)(ws + (6u << 20));  // ~5 MB (bf16 B)

    fbsnn_hidden<<<M_TOTAL, 256, 0, stream>>>(t, X, W1, b1, W2, b2, gc, Cb, u);
    fbsnn_build_pt<<<NPAD, 256, 0, stream>>>(W1, PTb);
    fbsnn_grad<<<M_TOTAL, 128, 0, stream>>>(gc, W1, DuDt, DuDx);
    fbsnn_hessian_wmma<<<dim3(NPAD / NBLK, M_TOTAL / MBLK), 256, 0, stream>>>(Cb, PTb, H);
}
